// MyLSTM_83253646065787
// MI455X (gfx1250) — compile-verified, matrix operand order checked by reading, not run on hardware
//
#include <hip/hip_runtime.h>

// ---------------------------------------------------------------------------
// 2-layer LSTM (B=4096, T=512, H=32, I=1) + final linear, fused.
// Workgroup = 64 threads = 2 waves, one 16-row batch tile:
//   wave 0: layer-0 recurrence  (v_wmma_f32_16x16x32_f16, 8/step)
//   wave 1: layer-1 recurrence  (16/step, K-chained input proj + recurrence,
//           C operand = inline 0, bias folded into v_tanh-based activations)
// Software pipeline: wave1 consumes h0[t] one step behind wave0 through an
// LDS double buffer; one workgroup barrier per step.
// ---------------------------------------------------------------------------

typedef __attribute__((ext_vector_type(16))) _Float16 v16h;
typedef __attribute__((ext_vector_type(8)))  _Float16 v8h;
typedef __attribute__((ext_vector_type(8)))  float    v8f;

#define TLEN 512
#define HID  32

// Native tanh trans-op (confirmed: lowers to v_tanh_f32); exp-based fallback.
#if defined(__has_builtin)
#if __has_builtin(__builtin_amdgcn_tanhf)
#define FAST_TANH(x) __builtin_amdgcn_tanhf(x)
#endif
#endif
#ifndef FAST_TANH
#define FAST_TANH(x) (1.0f - 2.0f / (__expf(2.0f * (x)) + 1.0f))
#endif

__device__ __forceinline__ float tanh_(float x)    { return FAST_TANH(x); }
__device__ __forceinline__ float sigmoid_(float x) { return 0.5f * FAST_TANH(0.5f * x) + 0.5f; }
// sigmoid(x + b) with hb = 0.5*b precomputed: one fma + one tanh + one fma.
__device__ __forceinline__ float sigmoid_b_(float x, float hb) {
    return 0.5f * FAST_TANH(0.5f * x + hb) + 0.5f;
}

__device__ __forceinline__ v8f wmma_f16(v16h a, v16h b, v8f c) {
    return __builtin_amdgcn_wmma_f32_16x16x32_f16(false, a, false, b,
                                                  (short)0, c, false, false);
}

// Build a per-lane B-fragment for W[g, 0:32] (row-major [128,32]).
// K map: lanes 0-15 -> K = 0..7,16..23 ; lanes 16-31 -> K = 8..15,24..31.
__device__ __forceinline__ v16h make_bfrag(const float* __restrict__ W,
                                           int g, int hs) {
    const float* r = W + g * HID + 8 * hs;
    float4 a0 = *(const float4*)(r + 0);
    float4 a1 = *(const float4*)(r + 4);
    float4 b0 = *(const float4*)(r + 16);
    float4 b1 = *(const float4*)(r + 20);
    v16h f;
    f[0]=(_Float16)a0.x; f[1]=(_Float16)a0.y; f[2]=(_Float16)a0.z; f[3]=(_Float16)a0.w;
    f[4]=(_Float16)a1.x; f[5]=(_Float16)a1.y; f[6]=(_Float16)a1.z; f[7]=(_Float16)a1.w;
    f[8]=(_Float16)b0.x; f[9]=(_Float16)b0.y; f[10]=(_Float16)b0.z; f[11]=(_Float16)b0.w;
    f[12]=(_Float16)b1.x; f[13]=(_Float16)b1.y; f[14]=(_Float16)b1.z; f[15]=(_Float16)b1.w;
    return f;
}

// Reload h (stored row-major [b][k] f16 in LDS) as a WMMA A-fragment.
__device__ __forceinline__ v16h load_afrag(const _Float16* hb, int m, int hs) {
    v8h lo = *(const v8h*)&hb[m * HID + 8 * hs];
    v8h hi = *(const v8h*)&hb[m * HID + 16 + 8 * hs];
    return __builtin_shufflevector(lo, hi,
        0,1,2,3,4,5,6,7,8,9,10,11,12,13,14,15);
}

__global__ __launch_bounds__(64)
void lstm2_wmma_kernel(const float* __restrict__ x,
                       const float* __restrict__ wih0,
                       const float* __restrict__ whh0,
                       const float* __restrict__ bih0,
                       const float* __restrict__ bhh0,
                       const float* __restrict__ wih1,
                       const float* __restrict__ whh1,
                       const float* __restrict__ bih1,
                       const float* __restrict__ bhh1,
                       const float* __restrict__ wlin,
                       const float* __restrict__ blin,
                       float* __restrict__ out)
{
    __shared__ __align__(16) float    xs[TLEN * 16];      // x transposed [t][b]
    __shared__ __align__(16) _Float16 h0buf[2][16 * HID]; // h0 double buffer
    __shared__ __align__(16) _Float16 h1buf[16 * HID];    // h1 round-trip

    const int tid  = threadIdx.x;
    const int wid  = tid >> 5;
    const int lane = tid & 31;
    const int m    = lane & 15;
    const int hs   = lane >> 4;
    const int b0   = blockIdx.x * 16;

    // ---- cooperative staging of x into LDS, transposed ----
    for (int b = 0; b < 16; ++b) {
        const float* xg = x + (size_t)(b0 + b) * TLEN;
        for (int t = tid; t < TLEN; t += 64)
            xs[t * 16 + b] = xg[t];
    }
    __syncthreads();

    if (wid == 0) {
        // ================= wave 0 : layer 0 =================
        v16h BL0[8];
        float w0c[8], bt0[8];
#pragma unroll
        for (int nt = 0; nt < 8; ++nt) {
            const int g = nt * 16 + m;
            BL0[nt] = make_bfrag(whh0, g, hs);
            w0c[nt] = wih0[g];
            bt0[nt] = bih0[g] + bhh0[g];
        }

        v16h h0f = {};
        v8f  c0lo = {}, c0hi = {};

        for (int t = 0; t < TLEN; ++t) {
            const float4* xp = (const float4*)&xs[t * 16 + 8 * hs];
            float4 xr0 = xp[0], xr1 = xp[1];
            float xrow[8] = {xr0.x, xr0.y, xr0.z, xr0.w,
                             xr1.x, xr1.y, xr1.z, xr1.w};
            _Float16* hb = h0buf[t & 1];

            auto l0half = [&](int kh, v8f& cc) {
                v8f ci, cf, cg, co;
#pragma unroll
                for (int v = 0; v < 8; ++v) {
                    ci[v] = xrow[v] * w0c[0 + kh] + bt0[0 + kh];
                    cf[v] = xrow[v] * w0c[2 + kh] + bt0[2 + kh];
                    cg[v] = xrow[v] * w0c[4 + kh] + bt0[4 + kh];
                    co[v] = xrow[v] * w0c[6 + kh] + bt0[6 + kh];
                }
                v8f ai = wmma_f16(h0f, BL0[0 + kh], ci);
                v8f af = wmma_f16(h0f, BL0[2 + kh], cf);
                v8f ag = wmma_f16(h0f, BL0[4 + kh], cg);
                v8f ao = wmma_f16(h0f, BL0[6 + kh], co);
#pragma unroll
                for (int v = 0; v < 8; ++v) {
                    float iv = sigmoid_(ai[v]);
                    float fv = sigmoid_(af[v]);
                    float gv = tanh_(ag[v]);
                    float ov = sigmoid_(ao[v]);
                    float c  = fv * cc[v] + iv * gv;
                    cc[v] = c;
                    float h  = ov * tanh_(c);
                    hb[(v + 8 * hs) * HID + (m + 16 * kh)] = (_Float16)h;
                }
            };
            l0half(0, c0lo);
            l0half(1, c0hi);

            asm volatile("s_wait_dscnt 0" ::: "memory");
            h0f = load_afrag(hb, m, hs);          // recurrent reload

            __syncthreads();                       // publish h0[t]
        }
        __syncthreads();                           // match wave1 final step
    } else {
        // ================= wave 1 : layer 1 =================
        v16h B1i[8], B1h[8];
        float bh1[8];   // 0.5 * (b_ih1 + b_hh1)   (sigmoid gates)
        float btg[8];   //        b_ih1 + b_hh1    (g gate)
#pragma unroll
        for (int nt = 0; nt < 8; ++nt) {
            const int g = nt * 16 + m;
            B1i[nt] = make_bfrag(wih1, g, hs);
            B1h[nt] = make_bfrag(whh1, g, hs);
            const float bsum = bih1[g] + bhh1[g];
            bh1[nt] = 0.5f * bsum;
            btg[nt] = bsum;
        }

        v16h h1f = {};
        v8f  c1lo = {}, c1hi = {};

        __syncthreads();                           // wait for h0[0]
        for (int t = 0; t < TLEN; ++t) {
            v16h h0f = load_afrag(h0buf[t & 1], m, hs);

            auto l1half = [&](int kh, v8f& cc) {
                const v8f z = {};                  // inline-0 C operand
                v8f ai = wmma_f16(h1f, B1h[0 + kh], z);
                v8f af = wmma_f16(h1f, B1h[2 + kh], z);
                v8f ag = wmma_f16(h1f, B1h[4 + kh], z);
                v8f ao = wmma_f16(h1f, B1h[6 + kh], z);
                ai = wmma_f16(h0f, B1i[0 + kh], ai);
                af = wmma_f16(h0f, B1i[2 + kh], af);
                ag = wmma_f16(h0f, B1i[4 + kh], ag);
                ao = wmma_f16(h0f, B1i[6 + kh], ao);
#pragma unroll
                for (int v = 0; v < 8; ++v) {
                    float iv = sigmoid_b_(ai[v], bh1[0 + kh]);
                    float fv = sigmoid_b_(af[v], bh1[2 + kh]);
                    float gv = tanh_(ag[v] + btg[4 + kh]);
                    float ov = sigmoid_b_(ao[v], bh1[6 + kh]);
                    float c  = fv * cc[v] + iv * gv;
                    cc[v] = c;
                    float h  = ov * tanh_(c);
                    h1buf[(v + 8 * hs) * HID + (m + 16 * kh)] = (_Float16)h;
                }
            };
            l1half(0, c1lo);
            l1half(1, c1hi);

            asm volatile("s_wait_dscnt 0" ::: "memory");
            h1f = load_afrag(h1buf, m, hs);        // recurrent reload

            __syncthreads();                       // release h0 buffer parity
        }

        // ---- final linear: out[b] = h1_last[b,:] . wlin + blin ----
        if (lane < 16) {
            float acc = blin[0];
#pragma unroll
            for (int k = 0; k < HID; ++k)
                acc += (float)h1buf[lane * HID + k] * wlin[k];
            out[b0 + lane] = acc;
        }
    }
}

extern "C" void kernel_launch(void* const* d_in, const int* in_sizes, int n_in,
                              void* d_out, int out_size, void* d_ws, size_t ws_size,
                              hipStream_t stream) {
    (void)n_in; (void)d_ws; (void)ws_size; (void)out_size;
    const float* x    = (const float*)d_in[0];
    const float* wih0 = (const float*)d_in[1];
    const float* whh0 = (const float*)d_in[2];
    const float* bih0 = (const float*)d_in[3];
    const float* bhh0 = (const float*)d_in[4];
    const float* wih1 = (const float*)d_in[5];
    const float* whh1 = (const float*)d_in[6];
    const float* bih1 = (const float*)d_in[7];
    const float* bhh1 = (const float*)d_in[8];
    const float* wlin = (const float*)d_in[9];
    const float* blin = (const float*)d_in[10];
    float* outp = (float*)d_out;

    const int Btot = in_sizes[0] / TLEN;   // 4096
    dim3 grid(Btot / 16), block(64);
    lstm2_wmma_kernel<<<grid, block, 0, stream>>>(
        x, wih0, whh0, bih0, bhh0, wih1, whh1, bih1, bhh1, wlin, blin, outp);
}
